// RelationAttentionHead_10539849744624
// MI455X (gfx1250) — matrix operand order repaired and checked by reading, not compiled, not run on hardware
//
#include <hip/hip_runtime.h>

typedef __attribute__((ext_vector_type(16))) _Float16 v16h;
typedef __attribute__((ext_vector_type(8)))  float    v8f;

#define EMB 128
#define HID 128

// ---------------------------------------------------------------------------
// CDNA5 async memory->LDS helpers (ASYNCcnt-tracked, per-lane 16B transfers)
// ---------------------------------------------------------------------------
__device__ __forceinline__ void async_load_b128(unsigned lds_off, const void* gptr)
{
    asm volatile("global_load_async_to_lds_b128 %0, %1, off"
                 :: "v"(lds_off), "v"((unsigned long long)(size_t)gptr)
                 : "memory");
}
__device__ __forceinline__ void wait_async0()
{
    asm volatile("s_wait_asynccnt 0x0" ::: "memory");
}
__device__ __forceinline__ void wait_async16()   // allow 16 newest outstanding
{
    asm volatile("s_wait_asynccnt 0x10" ::: "memory");
}

// ---------------------------------------------------------------------------
// Kernel 1: L2-normalize node embeddings, emit fp16 rows. One wave per node.
// ---------------------------------------------------------------------------
__global__ __launch_bounds__(256) void normalize_kernel(
    const float* __restrict__ xin, _Float16* __restrict__ x16, int N)
{
    const int wave = threadIdx.x >> 5;
    const int lane = threadIdx.x & 31;
    const int node = blockIdx.x * 8 + wave;
    if (node >= N) return;

    const float4 v = *(const float4*)(xin + (size_t)node * EMB + lane * 4);
    float ss = v.x * v.x + v.y * v.y + v.z * v.z + v.w * v.w;
    #pragma unroll
    for (int off = 16; off > 0; off >>= 1) ss += __shfl_xor(ss, off, 32);
    const float inv = 1.0f / fmaxf(sqrtf(ss), 1e-12f);

    union { _Float16 h[4]; uint2 u; } o;
    o.h[0] = (_Float16)(v.x * inv);
    o.h[1] = (_Float16)(v.y * inv);
    o.h[2] = (_Float16)(v.z * inv);
    o.h[3] = (_Float16)(v.w * inv);
    *(uint2*)(x16 + (size_t)node * EMB + lane * 4) = o.u;
}

// ---------------------------------------------------------------------------
// Kernel 2: precompute folded tables (single block).
//   bk2[d] = b_k[d] + sum_j b_edge[j]*W_k[j][d]   (b_edge folded out of GEMM)
//   bv2[d] = b_v[d] + sum_j b_edge[j]*W_v[j][d]
//   Qrel[R][n]      = rel_emb[R] @ W_q + b_q
//   WeT16[n][k]     = W_edge[k][n]
//   Arel16[R][j][h] = (sum_{d in h} W_k[j][d]*Qrel[R][d]) / sqrt(32)
//   crel[R][h]      = (sum_{d in h} bk2[d]  *Qrel[R][d]) / sqrt(32)
//   u[j][h]         =  sum_{d in h} W_v[j][d]*W_out[d]
//   beta[h]         =  sum_{d in h} bv2[d]  *W_out[d]
// ---------------------------------------------------------------------------
__global__ __launch_bounds__(256) void prep_kernel(
    const float* __restrict__ rel_emb, const float* __restrict__ W_edge,
    const float* __restrict__ b_edge,
    const float* __restrict__ W_q,     const float* __restrict__ b_q,
    const float* __restrict__ W_k,     const float* __restrict__ b_k,
    const float* __restrict__ W_v,     const float* __restrict__ b_v,
    const float* __restrict__ W_out,
    _Float16* __restrict__ WeT16, _Float16* __restrict__ Arel16,
    float* __restrict__ crel, float* __restrict__ uvec, float* __restrict__ beta)
{
    __shared__ float sQ[64 * HID];
    __shared__ float sBk2[HID];
    __shared__ float sBv2[HID];
    const int t = threadIdx.x;

    if (t < HID) {
        float accK = b_k[t], accV = b_v[t];
        for (int j = 0; j < HID; ++j) {
            accK += b_edge[j] * W_k[j * HID + t];
            accV += b_edge[j] * W_v[j * HID + t];
        }
        sBk2[t] = accK;
        sBv2[t] = accV;
    }
    for (int idx = t; idx < 64 * HID; idx += 256) {
        const int R = idx >> 7, n = idx & 127;
        float acc = b_q[n];
        for (int k = 0; k < 64; ++k) acc += rel_emb[R * 64 + k] * W_q[k * HID + n];
        sQ[idx] = acc;
    }
    __syncthreads();

    const float inv_scale = 0.17677669529663687f; // 1/sqrt(32)

    for (int idx = t; idx < HID * 256; idx += 256) {  // WeT16[n][k]
        const int n = idx >> 8, k = idx & 255;
        WeT16[idx] = (_Float16)W_edge[k * HID + n];
    }
    for (int idx = t; idx < 64 * HID * 4; idx += 256) {
        const int R = idx >> 9, j = (idx >> 2) & 127, h = idx & 3;
        float acc = 0.f;
        for (int d = 0; d < 32; ++d)
            acc += W_k[j * HID + h * 32 + d] * sQ[R * HID + h * 32 + d];
        Arel16[idx] = (_Float16)(acc * inv_scale);
    }
    if (t < 256) {
        const int R = t >> 2, h = t & 3;
        float acc = 0.f;
        for (int d = 0; d < 32; ++d)
            acc += sBk2[h * 32 + d] * sQ[R * HID + h * 32 + d];
        crel[t] = acc * inv_scale;
    }
    for (int idx = t; idx < 512; idx += 256) {
        const int j = idx >> 2, h = idx & 3;
        float acc = 0.f;
        for (int d = 0; d < 32; ++d)
            acc += W_v[j * HID + h * 32 + d] * W_out[h * 32 + d];
        uvec[idx] = acc;
    }
    if (t < 4) {
        float acc = 0.f;
        for (int d = 0; d < 32; ++d)
            acc += sBv2[t * 32 + d] * W_out[t * 32 + d];
        beta[t] = acc;
    }
}

// ---------------------------------------------------------------------------
// Kernel 3: fused per-edge pipeline. 4 waves/block, 16 edges per wave-tile,
// double-buffered async gather pipelined under the WMMA phase.
// ---------------------------------------------------------------------------
#define FEAT_ELEMS (16 * 256)                       // halves per feat buffer
#define SMEM_WET   0
#define SMEM_FEAT  (128 * 256 * 2)                  // 65536
#define SMEM_HBUF  (SMEM_FEAT + 4 * 2 * FEAT_ELEMS * 2)  // + 65536 (x2 buffers)
#define SMEM_F32   (SMEM_HBUF + 4 * 16 * 128 * 2)   // +16384
#define SMEM_TOTAL (SMEM_F32 + (512 + 256 + 4) * 4)

typedef union { v16h v; uint4 q[2]; } HRegs;

__global__ __launch_bounds__(128) void edge_attn_kernel(
    const _Float16* __restrict__ x16,        // [N,128] fp16 normalized
    const int*      __restrict__ edge_index, // [2,E]
    const int*      __restrict__ rel_type,   // [E]
    const _Float16* __restrict__ WeT16,      // [128,256]
    const _Float16* __restrict__ Arel16,     // [64,128,4]
    const float*    __restrict__ crel,       // [64,4]
    const float*    __restrict__ uvec,       // [128,4]
    const float*    __restrict__ beta,       // [4]
    const float*    __restrict__ b_out,      // [1]
    float*          __restrict__ out, int E, int numTiles)
{
    extern __shared__ char smem[];
    _Float16* sWeT     = (_Float16*)(smem + SMEM_WET);
    _Float16* sFeatAll = (_Float16*)(smem + SMEM_FEAT);
    _Float16* sHAll    = (_Float16*)(smem + SMEM_HBUF);
    float*    sU       = (float*)(smem + SMEM_F32);   // 512
    float*    sCrel    = sU + 512;                    // 256
    float*    sBeta    = sCrel + 256;                 // 4

    const int tid = threadIdx.x;
    { // fill 64KB weight table with async DMA; small tables via VGPRs
        const unsigned wbase = (unsigned)(size_t)sWeT;
        const char* g = (const char*)WeT16;
        for (int i = tid; i < 4096; i += 128)
            async_load_b128(wbase + i * 16, g + (size_t)i * 16);
        for (int i = tid; i < 512;  i += 128) sU[i] = uvec[i];
        for (int i = tid; i < 256;  i += 128) sCrel[i] = crel[i];
        if (tid < 4)   sBeta[tid]  = beta[tid];
        wait_async0();
    }
    __syncthreads();

    const int wave = tid >> 5;
    const int lane = tid & 31;
    const int r    = lane & 15;   // edge row within tile / N column
    const int part = lane >> 4;   // 0: src/low-K half, 1: tgt/high-K half
    _Float16* featB = sFeatAll + wave * (2 * FEAT_ELEMS);  // double-buffered
    _Float16* hbufT = sHAll    + wave * (16 * 128);        // [n][M] transposed
    const float bout = b_out[0];

    auto issue_gather = [&](int t, _Float16* dst) {
        int ee = t * 16 + r;
        ee = (ee < E) ? ee : (E - 1);
        const int nd = edge_index[part * E + ee];
        const unsigned lo = (unsigned)(size_t)(dst + r * 256 + part * 128);
        const char* gp = (const char*)(x16 + (size_t)nd * EMB);
        #pragma unroll
        for (int i = 0; i < 16; ++i)
            async_load_b128(lo + i * 16, gp + i * 16);
    };

    const int tstride = gridDim.x * 4;
    int tile = blockIdx.x * 4 + wave;
    int buf = 0;
    if (tile < numTiles) issue_gather(tile, featB);

    for (; tile < numTiles; tile += tstride, buf ^= 1) {
        // pipeline: kick off next tile's gather before consuming this one
        const int nt = tile + tstride;
        issue_gather((nt < numTiles) ? nt : tile, featB + (buf ^ 1) * FEAT_ELEMS);

        const int e      = tile * 16 + r;
        const bool valid = (e < E);
        const int eC     = valid ? e : (E - 1);
        const int rel    = rel_type[eC];
        const _Float16* arow = Arel16 + ((size_t)rel * 128 + part * 64) * 4;
        __builtin_prefetch(arow, 0, 1);          // warm A_rel row for attention
        __builtin_prefetch(arow + 256, 0, 1);

        const _Float16* feat = featB + buf * FEAT_ELEMS;
        wait_async16();   // in-order: <=16 outstanding => current buffer landed

        // ---- edge_hidden = feat @ W_edge (8 independent WMMA chains, zero C,
        //      B rotated through 3 register buffers, prefetch distance 2)
        v8f c[8];
        #pragma unroll
        for (int k0 = 0; k0 < 256; k0 += 32) {
            HRegs A;
            // A (16x32 f16): lanes<16: K=k0+0..7 & k0+16..23 ; lanes>=16: +8
            A.q[0] = *(const uint4*)(feat + r * 256 + k0 + part * 8);
            A.q[1] = *(const uint4*)(feat + r * 256 + k0 + 16 + part * 8);

            // B (32x16 f16): lanes<16: K=k0+0..7 & k0+16..23 ; lanes>=16: +8
            const _Float16* wb = sWeT + r * 256 + k0 + part * 8;
            HRegs B[3];
            B[0].q[0] = *(const uint4*)(wb);
            B[0].q[1] = *(const uint4*)(wb + 16);
            B[1].q[0] = *(const uint4*)(wb + 16 * 256);
            B[1].q[1] = *(const uint4*)(wb + 16 * 256 + 16);
            #pragma unroll
            for (int n0 = 0; n0 < 8; ++n0) {
                if (n0 + 2 < 8) {
                    HRegs& Bn = B[(n0 + 2) % 3];
                    Bn.q[0] = *(const uint4*)(wb + (n0 + 2) * 16 * 256);
                    Bn.q[1] = *(const uint4*)(wb + (n0 + 2) * 16 * 256 + 16);
                }
                if (k0 == 0) {
                    const v8f z = {0.f, 0.f, 0.f, 0.f, 0.f, 0.f, 0.f, 0.f};
                    c[n0] = __builtin_amdgcn_wmma_f32_16x16x32_f16(
                                false, A.v, false, B[n0 % 3].v, (short)0, z,
                                false, false);
                } else {
                    c[n0] = __builtin_amdgcn_wmma_f32_16x16x32_f16(
                                false, A.v, false, B[n0 % 3].v, (short)0, c[n0],
                                false, false);
                }
            }
        }
        // ---- store H transposed: lane's 8 consecutive M pack into one b128
        #pragma unroll
        for (int n0 = 0; n0 < 8; ++n0) {
            union { uint4 q; _Float16 h[8]; } o;
            #pragma unroll
            for (int i = 0; i < 8; ++i) o.h[i] = (_Float16)c[n0][i];
            *(uint4*)(hbufT + (n0 * 16 + r) * 16 + part * 8) = o.q;
        }

        // ---- attention: 2 lanes/edge, 64 hidden-elems each, 8 dot columns
        float sa0 = 0.f, sa1 = 0.f, sa2 = 0.f, sa3 = 0.f;
        float ta0 = 0.f, ta1 = 0.f, ta2 = 0.f, ta3 = 0.f;
        const int kst = part * 64;
        const float* up = sU + kst * 4;
        #pragma unroll 4
        for (int j = 0; j < 64; j += 2) {
            union { uint4 q; _Float16 h[8]; } aw;
            aw.q = *(const uint4*)(arow + j * 4);
            const float hv0 = (float)hbufT[(kst + j) * 16 + r];
            const float hv1 = (float)hbufT[(kst + j + 1) * 16 + r];
            const float4 uw0 = *(const float4*)(up + j * 4);
            const float4 uw1 = *(const float4*)(up + j * 4 + 4);
            sa0 += hv0 * (float)aw.h[0] + hv1 * (float)aw.h[4];
            sa1 += hv0 * (float)aw.h[1] + hv1 * (float)aw.h[5];
            sa2 += hv0 * (float)aw.h[2] + hv1 * (float)aw.h[6];
            sa3 += hv0 * (float)aw.h[3] + hv1 * (float)aw.h[7];
            ta0 += hv0 * uw0.x + hv1 * uw1.x;
            ta1 += hv0 * uw0.y + hv1 * uw1.y;
            ta2 += hv0 * uw0.z + hv1 * uw1.z;
            ta3 += hv0 * uw0.w + hv1 * uw1.w;
        }
        sa0 += __shfl_xor(sa0, 16, 32);  sa1 += __shfl_xor(sa1, 16, 32);
        sa2 += __shfl_xor(sa2, 16, 32);  sa3 += __shfl_xor(sa3, 16, 32);
        ta0 += __shfl_xor(ta0, 16, 32);  ta1 += __shfl_xor(ta1, 16, 32);
        ta2 += __shfl_xor(ta2, 16, 32);  ta3 += __shfl_xor(ta3, 16, 32);

        const float s0 = sa0 + sCrel[rel * 4 + 0];
        const float s1 = sa1 + sCrel[rel * 4 + 1];
        const float s2 = sa2 + sCrel[rel * 4 + 2];
        const float s3 = sa3 + sCrel[rel * 4 + 3];
        const float m  = fmaxf(fmaxf(s0, s1), fmaxf(s2, s3));
        const float w0 = __expf(s0 - m), w1 = __expf(s1 - m);
        const float w2 = __expf(s2 - m), w3 = __expf(s3 - m);
        const float wsum = w0 + w1 + w2 + w3;
        const float t0 = ta0 + sBeta[0], t1 = ta1 + sBeta[1];
        const float t2 = ta2 + sBeta[2], t3 = ta3 + sBeta[3];
        const float score = (w0 * t0 + w1 * t1 + w2 * t2 + w3 * t3) / wsum + bout;

        if (part == 0 && valid) out[e] = score;
    }
}

// ---------------------------------------------------------------------------
extern "C" void kernel_launch(void* const* d_in, const int* in_sizes, int n_in,
                              void* d_out, int out_size, void* d_ws, size_t ws_size,
                              hipStream_t stream)
{
    const float* node_emb  = (const float*)d_in[0];
    const int*   edge_idx  = (const int*)  d_in[1];
    const int*   rel_type  = (const int*)  d_in[2];
    const float* rel_emb   = (const float*)d_in[3];
    const float* W_edge    = (const float*)d_in[4];
    const float* b_edge    = (const float*)d_in[5];
    const float* W_q       = (const float*)d_in[6];
    const float* b_q       = (const float*)d_in[7];
    const float* W_k       = (const float*)d_in[8];
    const float* b_k       = (const float*)d_in[9];
    const float* W_v       = (const float*)d_in[10];
    const float* b_v       = (const float*)d_in[11];
    const float* W_out     = (const float*)d_in[12];
    const float* b_out     = (const float*)d_in[13];
    float* out = (float*)d_out;

    const int N = in_sizes[0] / EMB;
    const int E = in_sizes[1] / 2;

    char* ws = (char*)d_ws;
    size_t off = 0;
    auto alloc = [&](size_t bytes) -> void* {
        off = (off + 255) & ~(size_t)255;
        void* p = ws + off;
        off += bytes;
        return p;
    };
    _Float16* x16    = (_Float16*)alloc((size_t)N * EMB * sizeof(_Float16));
    _Float16* WeT16  = (_Float16*)alloc((size_t)HID * 256 * sizeof(_Float16));
    _Float16* Arel16 = (_Float16*)alloc((size_t)64 * HID * 4 * sizeof(_Float16));
    float*    crel   = (float*)alloc(256 * sizeof(float));
    float*    uvec   = (float*)alloc(512 * sizeof(float));
    float*    beta   = (float*)alloc(4 * sizeof(float));

    normalize_kernel<<<(N + 7) / 8, 256, 0, stream>>>(node_emb, x16, N);
    prep_kernel<<<1, 256, 0, stream>>>(rel_emb, W_edge, b_edge, W_q, b_q,
                                       W_k, b_k, W_v, b_v, W_out,
                                       WeT16, Arel16, crel, uvec, beta);

    const int numTiles = (E + 15) / 16;
    const int blocks = 1536;  // grid-stride; 2 WGs/WGP resident at ~147KB LDS
    edge_attn_kernel<<<blocks, 128, SMEM_TOTAL, stream>>>(
        x16, edge_idx, rel_type, WeT16, Arel16, crel, uvec, beta,
        b_out, out, E, numTiles);
}
